// GCN_15504831939241
// MI455X (gfx1250) — matrix operand order repaired and checked by reading, not compiled, not run on hardware
//
#include <hip/hip_runtime.h>
#include <hip/hip_bf16.h>

// ---------------------------------------------------------------------------
// GCN forward for MI455X (gfx1250, wave32, WMMA).
// Dense GEMMs run on v_wmma_f32_16x16x32_bf16 (f32 accumulate); scatter-add
// over edges uses global_atomic_add_f32 (L2-resident working set, 192MB L2);
// LayerNorm+ReLU fused per-node with wave32 shfl_xor reductions.
// ---------------------------------------------------------------------------

typedef __bf16 bf16;
typedef __attribute__((ext_vector_type(16))) __bf16 bf16x16;
typedef __attribute__((ext_vector_type(8)))  __bf16 bf16x8;
typedef __attribute__((ext_vector_type(4)))  __bf16 bf16x4;
typedef __attribute__((ext_vector_type(8)))  float  f32x8;

#define HDIM      128
#define G_GRAPHS  10
#define C_CLASSES 10

// ---------------------------------------------------------------- utilities
__global__ void fill_f32_kernel(float* __restrict__ p, float v, int n) {
    int i = blockIdx.x * blockDim.x + threadIdx.x;
    if (i < n) p[i] = v;
}

__global__ void cast_bf16_kernel(const float* __restrict__ src, bf16* __restrict__ dst, int n) {
    int i = blockIdx.x * blockDim.x + threadIdx.x;
    if (i < n) dst[i] = (bf16)src[i];
}

// Pack W [128x128 row-major f32] into per-wave WMMA B-fragment layout:
// Wp[((kt*8 + nt)*32 + lane)*16 + j] = bf16(W[k][n])
//   n = nt*16 + (lane&15);  k = kt*32 + (lane&16 ? 8 : 0) + (j&7) + (j&8 ? 16 : 0)
// so the GEMM kernel loads each lane's 16 bf16 B elements as contiguous 32B.
__global__ void pack_w_kernel(const float* __restrict__ W, bf16* __restrict__ Wp) {
    int idx = blockIdx.x * blockDim.x + threadIdx.x;
    if (idx >= 4 * 8 * 32 * 16) return;
    int j    = idx & 15;
    int lane = (idx >> 4) & 31;
    int nt   = (idx >> 9) & 7;
    int kt   = idx >> 12;
    int n = nt * 16 + (lane & 15);
    int k = kt * 32 + ((lane & 16) ? 8 : 0) + (j & 7) + ((j & 8) ? 16 : 0);
    Wp[idx] = (bf16)W[k * HDIM + n];
}

// deg[col]++ per edge (deg buffer pre-filled with 1.0 for the self-loop)
__global__ void deg_kernel(const int* __restrict__ cols, float* __restrict__ deg, int E) {
    int e = blockIdx.x * blockDim.x + threadIdx.x;
    if (e < E) atomicAdd(&deg[cols[e]], 1.0f);
}

__global__ void dis_kernel(float* __restrict__ deg, int n) {
    int i = blockIdx.x * blockDim.x + threadIdx.x;
    if (i < n) deg[i] = rsqrtf(deg[i]);   // deg >= 1 always (self-loop)
}

// ------------------------------------------------------------ WMMA GEMM
// C[M x 128] = A[M x 128](bf16) @ Wpacked(bf16), f32 accumulate.
// One wave per 16x16 output tile; K=128 -> 4 x v_wmma_f32_16x16x32_bf16.
__global__ void gemm_wmma_kernel(const bf16* __restrict__ A,
                                 const bf16* __restrict__ Wp,
                                 float* __restrict__ C, int M) {
    int gtid = blockIdx.x * blockDim.x + threadIdx.x;
    int wave = gtid >> 5;
    int lane = threadIdx.x & 31;
    int mt = wave >> 3;            // HDIM/16 == 8 n-tiles per m-tile
    int nt = wave & 7;
    if (mt * 16 >= M) return;      // wave-uniform: EXEC stays all-ones for WMMA

    int row = mt * 16 + (lane & 15);
    if (row >= M) row = M - 1;     // tail clamp (store is guarded below)
    int koA = (lane & 16) ? 8 : 0; // A: lanes 0-15 hold K 0-7/16-23, 16-31 hold K 8-15/24-31
    const bf16* arow = A + (size_t)row * HDIM;

    f32x8 acc = {};
    union Frag { bf16x16 v; bf16x8 h[2]; };
    #pragma unroll
    for (int kt = 0; kt < 4; ++kt) {
        Frag a, b;
        int k0 = kt * 32 + koA;
        a.h[0] = *(const bf16x8*)(arow + k0);        // K = k0..k0+7     (16B load)
        a.h[1] = *(const bf16x8*)(arow + k0 + 16);   // K = k0+16..k0+23 (16B load)
        const bf16* wp = Wp + (size_t)(((kt * 8 + nt) * 32 + lane) * 16);
        b.h[0] = *(const bf16x8*)(wp);
        b.h[1] = *(const bf16x8*)(wp + 8);
        acc = __builtin_amdgcn_wmma_f32_16x16x32_bf16(
            false, a.v, false, b.v, (short)0, acc, false, false);
    }

    // C layout: VGPR r, lane L -> row = r + 8*(L>=16), col = L&15
    int crow = mt * 16 + ((lane & 16) ? 8 : 0);
    int ccol = nt * 16 + (lane & 15);
    float* cp = C + (size_t)crow * HDIM + ccol;
    if (mt * 16 + 16 <= M) {
        // full tile (always taken when M % 16 == 0): straight-line stores
        #pragma unroll
        for (int r = 0; r < 8; ++r) cp[(size_t)r * HDIM] = acc[r];
    } else {
        #pragma unroll
        for (int r = 0; r < 8; ++r)
            if (crow + r < M) cp[(size_t)r * HDIM] = acc[r];
    }
}

// --------------------------------------------------- edge scatter-aggregate
// agg[col] += h[row] * dis[row]*dis[col]; one wave per edge, float4 per lane.
__global__ void scatter_kernel(const int* __restrict__ rows, const int* __restrict__ cols,
                               const float* __restrict__ dis, const float* __restrict__ h,
                               float* __restrict__ agg, int E) {
    long long t = (long long)blockIdx.x * blockDim.x + threadIdx.x;
    int lane = (int)(t & 31);
    long long e = t >> 5;
    if (e >= E) return;
    int r = rows[e], c = cols[e];
    float nrm = dis[r] * dis[c];
    const float4 hv = ((const float4*)(h + (size_t)r * HDIM))[lane];
    float* ap = agg + (size_t)c * HDIM + lane * 4;
    atomicAdd(ap + 0, hv.x * nrm);
    atomicAdd(ap + 1, hv.y * nrm);
    atomicAdd(ap + 2, hv.z * nrm);
    atomicAdd(ap + 3, hv.w * nrm);
}

// ----------------- fused: self-loop + bias + LayerNorm + ReLU (+bf16/+pool)
__global__ void fused_ln_kernel(const float* __restrict__ h, const float* __restrict__ dis,
                                const float* __restrict__ bias, const float* __restrict__ gamma,
                                const float* __restrict__ beta, const float* __restrict__ agg,
                                bf16* __restrict__ out_bf,
                                float* __restrict__ pooled, float* __restrict__ cnt,
                                const int* __restrict__ batch, int n) {
    long long t = (long long)blockIdx.x * blockDim.x + threadIdx.x;
    int lane = (int)(t & 31);
    long long i = t >> 5;
    if (i >= n) return;

    float dsq = dis[i]; dsq *= dsq;                       // self-loop norm
    const float4 hv = ((const float4*)(h   + (size_t)i * HDIM))[lane];
    const float4 av = ((const float4*)(agg + (size_t)i * HDIM))[lane];
    const float4 bv = ((const float4*)bias)[lane];
    const float4 gv = ((const float4*)gamma)[lane];
    const float4 tv = ((const float4*)beta)[lane];
    int c0 = lane * 4;

    float v0 = av.x + hv.x * dsq + bv.x;
    float v1 = av.y + hv.y * dsq + bv.y;
    float v2 = av.z + hv.z * dsq + bv.z;
    float v3 = av.w + hv.w * dsq + bv.w;

    float s = v0 + v1 + v2 + v3;
    #pragma unroll
    for (int off = 16; off; off >>= 1) s += __shfl_xor(s, off);
    float mu = s * (1.0f / HDIM);

    float e0 = v0 - mu, e1 = v1 - mu, e2 = v2 - mu, e3 = v3 - mu;
    float q = e0 * e0 + e1 * e1 + e2 * e2 + e3 * e3;
    #pragma unroll
    for (int off = 16; off; off >>= 1) q += __shfl_xor(q, off);
    float rstd = rsqrtf(q * (1.0f / HDIM) + 1e-5f);

    float y0 = fmaxf(e0 * rstd * gv.x + tv.x, 0.0f);
    float y1 = fmaxf(e1 * rstd * gv.y + tv.y, 0.0f);
    float y2 = fmaxf(e2 * rstd * gv.z + tv.z, 0.0f);
    float y3 = fmaxf(e3 * rstd * gv.w + tv.w, 0.0f);

    if (out_bf) {                                         // layer-2 GEMM input
        bf16x4 o;
        o.x = (bf16)y0; o.y = (bf16)y1; o.z = (bf16)y2; o.w = (bf16)y3;
        *(bf16x4*)(out_bf + (size_t)i * HDIM + c0) = o;   // one 8B store
    }
    if (pooled) {                                         // global mean pool (sums+counts)
        int b = batch[i];
        float* pp = pooled + (size_t)b * HDIM + c0;
        atomicAdd(pp + 0, y0); atomicAdd(pp + 1, y1);
        atomicAdd(pp + 2, y2); atomicAdd(pp + 3, y3);
        if (lane == 0) atomicAdd(&cnt[b], 1.0f);
    }
}

// -------------------------------------------------- tiny head: pooled@Wl+bl
__global__ void final_kernel(const float* __restrict__ pooled, const float* __restrict__ cnt,
                             const float* __restrict__ Wl, const float* __restrict__ bl,
                             float* __restrict__ out) {
    int t = threadIdx.x;
    if (t >= G_GRAPHS * C_CLASSES) return;
    int g = t / C_CLASSES, c = t % C_CLASSES;
    float inv = 1.0f / fmaxf(cnt[g], 1.0f);
    float s = bl[c];
    #pragma unroll 4
    for (int k = 0; k < HDIM; ++k)
        s += pooled[g * HDIM + k] * inv * Wl[k * C_CLASSES + c];
    out[t] = s;
}

// ---------------------------------------------------------------- launcher
extern "C" void kernel_launch(void* const* d_in, const int* in_sizes, int n_in,
                              void* d_out, int out_size, void* d_ws, size_t ws_size,
                              hipStream_t stream) {
    (void)n_in; (void)out_size; (void)ws_size;

    const float* x     = (const float*)d_in[0];
    const int*   ei    = (const int*)  d_in[1];   // edge_index [2, E]
    const int*   batch = (const int*)  d_in[2];
    const float* W1    = (const float*)d_in[3];
    const float* b1    = (const float*)d_in[4];
    const float* g1    = (const float*)d_in[5];
    const float* be1   = (const float*)d_in[6];
    const float* W2    = (const float*)d_in[7];
    const float* b2    = (const float*)d_in[8];
    const float* g2    = (const float*)d_in[9];
    const float* be2   = (const float*)d_in[10];
    const float* Wl    = (const float*)d_in[11];
    const float* bl    = (const float*)d_in[12];
    float* out = (float*)d_out;

    const int N = in_sizes[0] / HDIM;
    const int E = in_sizes[1] / 2;
    const int* rows = ei;
    const int* cols = ei + E;

    // workspace carve-out (256B aligned); total ~77MB
    char* ws = (char*)d_ws;
    size_t off = 0;
    auto take = [&](size_t bytes) {
        char* p = ws + off;
        off = (off + bytes + 255) & ~(size_t)255;
        return p;
    };
    float* dis    = (float*)take((size_t)N * 4);
    bf16*  xb     = (bf16*) take((size_t)N * HDIM * 2);
    bf16*  hb     = (bf16*) take((size_t)N * HDIM * 2);
    float* h      = (float*)take((size_t)N * HDIM * 4);
    float* agg    = (float*)take((size_t)N * HDIM * 4);
    bf16*  Wpk1   = (bf16*) take((size_t)HDIM * HDIM * 2);
    bf16*  Wpk2   = (bf16*) take((size_t)HDIM * HDIM * 2);
    float* pooled = (float*)take((size_t)G_GRAPHS * HDIM * 4);
    float* cnt    = (float*)take((size_t)G_GRAPHS * 4);

    const int B = 256;
    auto cdiv = [](long long a, long long b) { return (unsigned)((a + b - 1) / b); };
    const long long NH = (long long)N * HDIM;
    const unsigned gemmBlocks = cdiv((long long)((N + 15) / 16) * 8 * 32, B);

    // init
    fill_f32_kernel<<<cdiv(N, B), B, 0, stream>>>(dis, 1.0f, N);           // self-loop degree
    fill_f32_kernel<<<cdiv(NH, B), B, 0, stream>>>(agg, 0.0f, (int)NH);
    fill_f32_kernel<<<cdiv(G_GRAPHS * HDIM, B), B, 0, stream>>>(pooled, 0.0f, G_GRAPHS * HDIM);
    fill_f32_kernel<<<1, 32, 0, stream>>>(cnt, 0.0f, G_GRAPHS);

    // precompute: bf16 features, packed weights, symmetric norm
    cast_bf16_kernel<<<cdiv(NH, B), B, 0, stream>>>(x, xb, (int)NH);
    pack_w_kernel<<<cdiv(16384, B), B, 0, stream>>>(W1, Wpk1);
    pack_w_kernel<<<cdiv(16384, B), B, 0, stream>>>(W2, Wpk2);
    deg_kernel<<<cdiv(E, B), B, 0, stream>>>(cols, dis, E);
    dis_kernel<<<cdiv(N, B), B, 0, stream>>>(dis, N);

    // layer 1: GEMM -> scatter -> fused self-loop+bias+LN+ReLU (emit bf16)
    gemm_wmma_kernel<<<gemmBlocks, B, 0, stream>>>(xb, Wpk1, h, N);
    scatter_kernel<<<cdiv((long long)E * 32, B), B, 0, stream>>>(rows, cols, dis, h, agg, E);
    fused_ln_kernel<<<cdiv((long long)N * 32, B), B, 0, stream>>>(
        h, dis, b1, g1, be1, agg, hb, nullptr, nullptr, nullptr, N);

    // layer 2: same, fused kernel also does mean-pool accumulation
    fill_f32_kernel<<<cdiv(NH, B), B, 0, stream>>>(agg, 0.0f, (int)NH);
    gemm_wmma_kernel<<<gemmBlocks, B, 0, stream>>>(hb, Wpk2, h, N);
    scatter_kernel<<<cdiv((long long)E * 32, B), B, 0, stream>>>(rows, cols, dis, h, agg, E);
    fused_ln_kernel<<<cdiv((long long)N * 32, B), B, 0, stream>>>(
        h, dis, b2, g2, be2, agg, nullptr, pooled, cnt, batch, N);

    // head
    final_kernel<<<1, 128, 0, stream>>>(pooled, cnt, Wl, bl, out);
}
